// SMARTAgentDecoder_79894981640301
// MI455X (gfx1250) — matrix operand order repaired and checked by reading, not compiled, not run on hardware
//
#include <hip/hip_runtime.h>
#include <hip/hip_bf16.h>
#include <math.h>

#define NAGT 1024
#define NSTP 18
#define NSN  18432      // NAGT*NSTP
#define NPL  8192
#define NPLS 147456     // NSTP*NPL
#define NTOK 2048

#define CDIV(a,b) (((a)+(b)-1)/(b))
#define PI_F 3.14159265358979323846f
#define TWO_PI_F 6.28318530717958647692f

typedef __attribute__((ext_vector_type(16))) __bf16 bf16x16;
typedef __attribute__((ext_vector_type(8)))  __bf16 bf16x8;
typedef __attribute__((ext_vector_type(4)))  __bf16 bf16x4;
typedef __attribute__((ext_vector_type(8)))  float  f32x8;

__device__ __forceinline__ float wrapf(float a) {
  float r = fmodf(a + PI_F, TWO_PI_F);
  if (r < 0.f) r += TWO_PI_F;
  return r - PI_F;
}

// ---------------------------------------------------------------------------
// Weight pre-convert: W[K,N] f32 -> bf16 tiles in B-fragment order.
// Layout: wbf[((kt*N)+n)*32 + kk]  with k = kt*32+kk, zero-padded for k>=K.
// ---------------------------------------------------------------------------
__global__ void k_w2bf(const float* __restrict__ W, __bf16* __restrict__ out,
                       int K, int N, int KT)
{
  size_t t = (size_t)blockIdx.x * blockDim.x + threadIdx.x;
  size_t total = (size_t)KT * 32 * N;
  if (t >= total) return;
  int n  = (int)(t % N);
  int k  = (int)(t / N);          // 0 .. KT*32-1
  int kt = k >> 5, kk = k & 31;
  float f = (k < K) ? W[(size_t)k * N + n] : 0.f;
  out[((size_t)kt * N + n) * 32 + kk] = (__bf16)f;
}

// ---------------------------------------------------------------------------
// Generic GEMM: C[M,N] = act( A[M,K] @ W[K,N] + bias + add (+C if acc) )
// Block tile 64x128, 8 waves, four 16x16 WMMA tiles per wave, K step 32.
//  - B fragments: two aligned global_load_b128 per wave per K-step from
//    preconverted bf16 weights (L2-resident), amortized over 4 WMMAs.
//  - A tile: double-buffered LDS staging (bf16), 2 float4 loads +
//    1 ds_store_b128 per thread, single barrier per K-step.
// ---------------------------------------------------------------------------
__global__ __launch_bounds__(256)
void k_gemm(const float* __restrict__ A, const __bf16* __restrict__ Wbf,
            const float* __restrict__ bias, const float* __restrict__ add,
            float* __restrict__ C, int M, int N, int K, int KT, int lda,
            int accflag, int act, int safeK)
{
  __shared__ __bf16 As[2][64 * 40];
  const int t    = threadIdx.x;
  const int wv   = t >> 5;
  const int lane = t & 31;
  const int row0 = blockIdx.x * 64;
  const int col0 = blockIdx.y * 128;

  const int mA  = lane & 15;
  const int kbA = (lane & 16) ? 8 : 0;
  const int nB  = (wv << 4) | (lane & 15);
  const int kbB = (lane & 16) ? 8 : 0;

  const int arow = t & 63;        // A staging: row
  const int akh  = t >> 6;        // A staging: k group (4 groups of 8)

  const bool allfast = ((lda & 3) == 0) && (row0 + 64 <= M) &&
                       (safeK || ((K & 31) == 0));

  f32x8 acc[4] = {{}, {}, {}, {}};

  auto stage = [&](int kt, int buf) {
    const int k0 = kt << 5;
    const int gr = row0 + arow;
    const float* ap = A + (size_t)gr * lda + k0 + akh * 8;
    bf16x8 pa;
    if (allfast) {
      float4 f0 = *(const float4*)ap;
      float4 f1 = *(const float4*)(ap + 4);
      pa[0] = (__bf16)f0.x; pa[1] = (__bf16)f0.y;
      pa[2] = (__bf16)f0.z; pa[3] = (__bf16)f0.w;
      pa[4] = (__bf16)f1.x; pa[5] = (__bf16)f1.y;
      pa[6] = (__bf16)f1.z; pa[7] = (__bf16)f1.w;
    } else {
#pragma unroll
      for (int i = 0; i < 8; ++i) {
        int gk = k0 + akh * 8 + i;
        pa[i] = (__bf16)((gr < M && gk < K) ? ap[i] : 0.f);
      }
    }
    *(bf16x8*)&As[buf][arow * 40 + akh * 8] = pa;
    if (allfast) __builtin_prefetch(ap + 64, 0, 1);
  };

  stage(0, 0);
  __syncthreads();

  for (int kt = 0; kt < KT; ++kt) {
    if (kt + 1 < KT) stage(kt + 1, (kt + 1) & 1);

    const __bf16* asb = As[kt & 1];
    const bf16x8* bp = (const bf16x8*)(Wbf + ((size_t)kt * N + col0 + nB) * 32 + kbB);
    bf16x8 b0 = bp[0];
    bf16x8 b1 = bp[2];
    bf16x16 bfr;
#pragma unroll
    for (int i = 0; i < 8; ++i) { bfr[i] = b0[i]; bfr[i + 8] = b1[i]; }

#pragma unroll
    for (int m = 0; m < 4; ++m) {
      bf16x8 alo = *(const bf16x8*)&asb[(m * 16 + mA) * 40 + kbA];
      bf16x8 ahi = *(const bf16x8*)&asb[(m * 16 + mA) * 40 + 16 + kbA];
      bf16x16 a;
#pragma unroll
      for (int i = 0; i < 8; ++i) { a[i] = alo[i]; a[i + 8] = ahi[i]; }
      acc[m] = __builtin_amdgcn_wmma_f32_16x16x32_bf16(false, a, false, bfr,
                                                       (short)0, acc[m], false, false);
    }
    __syncthreads();
  }

  const int colw = col0 + nB;
  const int rsel = (lane & 16) ? 8 : 0;
  const float bvv = bias ? bias[colw] : 0.f;

#pragma unroll
  for (int m = 0; m < 4; ++m) {
#pragma unroll
    for (int r = 0; r < 8; ++r) {
      int grow = row0 + m * 16 + r + rsel;
      if (grow < M) {
        size_t o = (size_t)grow * N + colw;
        float v = acc[m][r] + bvv;
        if (add) v += add[o];
        if (accflag) v += C[o];
        if (act == 1) v = fmaxf(v, 0.f);
        else if (act == 2) v = 1.f / (1.f + expf(-v));
        C[o] = v;
      }
    }
  }
}

// ---------------------------------------------------------------------------
// LayerNorm over 128 cols, one wave32 per row (4 elems/lane), optional relu.
// ---------------------------------------------------------------------------
__global__ __launch_bounds__(256)
void k_ln128(const float* __restrict__ x, const float* __restrict__ g,
             const float* __restrict__ b, float* __restrict__ y, int M, int relu)
{
  int row  = blockIdx.x * 8 + (threadIdx.x >> 5);
  int lane = threadIdx.x & 31;
  if (row >= M) return;
  const float* xr = x + (size_t)row * 128;
  float v[4];
  float s = 0.f;
#pragma unroll
  for (int i = 0; i < 4; ++i) { v[i] = xr[lane + 32 * i]; s += v[i]; }
#pragma unroll
  for (int o = 16; o > 0; o >>= 1) s += __shfl_xor(s, o, 32);
  float mu = s * (1.f / 128.f);
  float s2 = 0.f;
#pragma unroll
  for (int i = 0; i < 4; ++i) { float d = v[i] - mu; s2 += d * d; }
#pragma unroll
  for (int o = 16; o > 0; o >>= 1) s2 += __shfl_xor(s2, o, 32);
  float rstd = rsqrtf(s2 * (1.f / 128.f) + 1e-5f);
  float* yr = y + (size_t)row * 128;
#pragma unroll
  for (int i = 0; i < 4; ++i) {
    int c = lane + 32 * i;
    float o = (v[i] - mu) * rstd * g[c] + b[c];
    if (relu) o = fmaxf(o, 0.f);
    yr[c] = o;
  }
}

// ---------------------------------------------------------------------------
// Small elementwise / gather / scatter kernels
// ---------------------------------------------------------------------------
__global__ void k_fill(float* __restrict__ p, float v, size_t n) {
  size_t i = (size_t)blockIdx.x * blockDim.x + threadIdx.x;
  if (i < n) p[i] = v;
}

__global__ void k_gather_emb(const float* __restrict__ tables, const int* __restrict__ tok_idx,
                             const int* __restrict__ atype, float* __restrict__ emb)
{
  size_t t = (size_t)blockIdx.x * blockDim.x + threadIdx.x;
  if (t >= (size_t)NSN * 128) return;
  int r = (int)(t >> 7), d = (int)(t & 127);
  int a = r / NSTP;
  int ty = atype[a];
  int tk = tok_idx[r];
  emb[t] = tables[((size_t)ty * NTOK + tk) * 128 + d];
}

__global__ void k_typeadd(float* __restrict__ cat_a, const float* __restrict__ type_table,
                          const int* __restrict__ atype)
{
  size_t t = (size_t)blockIdx.x * blockDim.x + threadIdx.x;
  if (t >= (size_t)NAGT * 128) return;
  int a = (int)(t >> 7);
  cat_a[t] += type_table[(size_t)atype[a] * 128 + (t & 127)];
}

__global__ void k_featcont(const float* __restrict__ pos, const float* __restrict__ head,
                           float* __restrict__ out)
{
  int i = blockIdx.x * blockDim.x + threadIdx.x;
  if (i >= NSN) return;
  int s = i % NSTP;
  float mx = 0.f, my = 0.f;
  if (s > 0) {
    mx = pos[2 * i]     - pos[2 * (i - 1)];
    my = pos[2 * i + 1] - pos[2 * (i - 1) + 1];
  }
  float h = head[i];
  float cx = cosf(h), cy = sinf(h);
  out[2 * i]     = sqrtf(mx * mx + my * my);
  out[2 * i + 1] = atan2f(cx * my - cy * mx, cx * mx + cy * my);
}

// fourier feature matrix, stride 132 (16B-aligned rows), cols 129..131 zeroed
__global__ void k_ffeat(const float* __restrict__ x, int C, int c,
                        const float* __restrict__ freqs, float* __restrict__ feat, int N)
{
  size_t t = (size_t)blockIdx.x * blockDim.x + threadIdx.x;
  if (t >= (size_t)N * 132) return;
  int r = (int)(t / 132), j = (int)(t % 132);
  float xv = x[(size_t)r * C + c];
  float o;
  if (j < 64)       o = cosf(xv * freqs[j] * TWO_PI_F);
  else if (j < 128) o = sinf(xv * freqs[j - 64] * TWO_PI_F);
  else if (j == 128) o = xv;
  else              o = 0.f;
  feat[t] = o;
}

__global__ void k_addagent(float* __restrict__ out, const float* __restrict__ cat_a, int N)
{
  size_t t = (size_t)blockIdx.x * blockDim.x + threadIdx.x;
  if (t >= (size_t)N * 128) return;
  int r = (int)(t >> 7);
  out[t] += cat_a[(size_t)(r / NSTP) * 128 + (t & 127)];
}

__global__ void k_concat2(float* __restrict__ dst, const float* __restrict__ A,
                          const float* __restrict__ B, int M)
{
  size_t t = (size_t)blockIdx.x * blockDim.x + threadIdx.x;
  if (t >= (size_t)M * 256) return;
  int r = (int)(t >> 8), d = (int)(t & 255);
  dst[t] = (d < 128) ? A[(size_t)r * 128 + d] : B[(size_t)r * 128 + (d - 128)];
}

__global__ void k_tile_map(float* __restrict__ map_s, const float* __restrict__ mf)
{
  size_t t = (size_t)blockIdx.x * blockDim.x + threadIdx.x;
  if (t >= (size_t)NPLS * 128) return;
  int r = (int)(t >> 7);
  int p = r % NPL;
  map_s[t] = mf[((size_t)p << 7) | (t & 127)];
}

// dir 0: (a,s)-major -> (s,a)-major ; dir 1: reverse
__global__ void k_transpose(float* __restrict__ dst, const float* __restrict__ src, int dir)
{
  size_t t = (size_t)blockIdx.x * blockDim.x + threadIdx.x;
  if (t >= (size_t)NSN * 128) return;
  int r = (int)(t >> 7), d = (int)(t & 127);
  int ro;
  if (dir == 0) { int a = r / NSTP, s = r % NSTP; ro = (s << 10) + a; }
  else          { int s = r >> 10, a = r & 1023;  ro = a * NSTP + s; }
  dst[(size_t)ro * 128 + d] = src[t];
}

__global__ void k_rel_t(const float* __restrict__ pos, const float* __restrict__ head,
                        const int* __restrict__ es, const int* __restrict__ ed,
                        float* __restrict__ out, int E)
{
  int e = blockIdx.x * blockDim.x + threadIdx.x;
  if (e >= E) return;
  int s = es[e], d = ed[e];
  float px = pos[2 * s] - pos[2 * d];
  float py = pos[2 * s + 1] - pos[2 * d + 1];
  float hd = head[d];
  float cx = cosf(hd), cy = sinf(hd);
  out[(size_t)e * 4 + 0] = sqrtf(px * px + py * py);
  out[(size_t)e * 4 + 1] = atan2f(cx * py - cy * px, cx * px + cy * py);
  out[(size_t)e * 4 + 2] = wrapf(head[s] - hd);
  out[(size_t)e * 4 + 3] = (float)(s - d);
}

__global__ void k_rel_a2a(const float* __restrict__ pos, const float* __restrict__ head,
                          const int* __restrict__ es, const int* __restrict__ ed,
                          float* __restrict__ out, int E)
{
  int e = blockIdx.x * blockDim.x + threadIdx.x;
  if (e >= E) return;
  int is = es[e], id_ = ed[e];
  int as_ = is & 1023, ss = is >> 10;
  int ad  = id_ & 1023, sd = id_ >> 10;
  int ns = as_ * NSTP + ss, nd = ad * NSTP + sd;
  float px = pos[2 * ns] - pos[2 * nd];
  float py = pos[2 * ns + 1] - pos[2 * nd + 1];
  float hd = head[nd];
  float cx = cosf(hd), cy = sinf(hd);
  out[(size_t)e * 3 + 0] = sqrtf(px * px + py * py);
  out[(size_t)e * 3 + 1] = atan2f(cx * py - cy * px, cx * px + cy * py);
  out[(size_t)e * 3 + 2] = wrapf(head[ns] - hd);
}

__global__ void k_rel_pl2a(const float* __restrict__ mpos, const float* __restrict__ morient,
                           const float* __restrict__ pos, const float* __restrict__ head,
                           const int* __restrict__ es, const int* __restrict__ ed,
                           float* __restrict__ out, int E)
{
  int e = blockIdx.x * blockDim.x + threadIdx.x;
  if (e >= E) return;
  int pl = es[e] & (NPL - 1);
  int id_ = ed[e];
  int ad = id_ & 1023, sd = id_ >> 10;
  int nd = ad * NSTP + sd;
  float px = mpos[2 * pl] - pos[2 * nd];
  float py = mpos[2 * pl + 1] - pos[2 * nd + 1];
  float hd = head[nd];
  float cx = cosf(hd), cy = sinf(hd);
  out[(size_t)e * 3 + 0] = sqrtf(px * px + py * py);
  out[(size_t)e * 3 + 1] = atan2f(cx * py - cy * px, cx * px + cy * py);
  out[(size_t)e * 3 + 2] = wrapf(morient[pl] - hd);
}

// ---------------------------------------------------------------------------
// Segment softmax + aggregation (atomics)
// ---------------------------------------------------------------------------
__global__ void k_sim(const float* __restrict__ q, const float* __restrict__ k,
                      const float* __restrict__ rk, const int* __restrict__ es,
                      const int* __restrict__ ed, float* __restrict__ sim, int E)
{
  size_t t = (size_t)blockIdx.x * blockDim.x + threadIdx.x;
  if (t >= (size_t)E * 8) return;
  int e = (int)(t >> 3), h = (int)(t & 7);
  const float4* qp = (const float4*)(q  + (size_t)ed[e] * 128 + h * 16);
  const float4* kp = (const float4*)(k  + (size_t)es[e] * 128 + h * 16);
  const float4* rp = (const float4*)(rk + (size_t)e * 128 + h * 16);
  float s = 0.f;
#pragma unroll
  for (int i = 0; i < 4; ++i) {
    float4 qv = qp[i], kv = kp[i], rv = rp[i];
    s += qv.x * (kv.x + rv.x) + qv.y * (kv.y + rv.y)
       + qv.z * (kv.z + rv.z) + qv.w * (kv.w + rv.w);
  }
  sim[t] = s * 0.25f;
}

__device__ __forceinline__ void atomicMaxF(float* addr, float val) {
  int* ai = (int*)addr;
  int cur = __float_as_int(*addr);
  while (__int_as_float(cur) < val) {
    int prev = atomicCAS(ai, cur, __float_as_int(val));
    if (prev == cur) break;
    cur = prev;
  }
}

__global__ void k_segmax(const float* __restrict__ sim, const int* __restrict__ ed,
                         float* __restrict__ smax, int E)
{
  size_t t = (size_t)blockIdx.x * blockDim.x + threadIdx.x;
  if (t >= (size_t)E * 8) return;
  int e = (int)(t >> 3), h = (int)(t & 7);
  atomicMaxF(&smax[(size_t)ed[e] * 8 + h], sim[t]);
}

__global__ void k_expden(float* __restrict__ sim, const int* __restrict__ ed,
                         const float* __restrict__ smax, float* __restrict__ den, int E)
{
  size_t t = (size_t)blockIdx.x * blockDim.x + threadIdx.x;
  if (t >= (size_t)E * 8) return;
  int e = (int)(t >> 3), h = (int)(t & 7);
  float ex = expf(sim[t] - smax[(size_t)ed[e] * 8 + h]);
  sim[t] = ex;
  atomicAdd(&den[(size_t)ed[e] * 8 + h], ex);
}

__global__ void k_agg(const float* __restrict__ ex, const float* __restrict__ den,
                      const float* __restrict__ v, const float* __restrict__ rv,
                      const int* __restrict__ es, const int* __restrict__ ed,
                      float* __restrict__ agg, int E)
{
  size_t t = (size_t)blockIdx.x * blockDim.x + threadIdx.x;
  if (t >= (size_t)E * 128) return;
  int e = (int)(t >> 7), d = (int)(t & 127), h = d >> 4;
  int s = es[e], dd = ed[e];
  float alpha = ex[(size_t)e * 8 + h] / den[(size_t)dd * 8 + h];
  atomicAdd(&agg[(size_t)dd * 128 + d], alpha * (v[(size_t)s * 128 + d] + rv[t]));
}

__global__ void k_upd(float* __restrict__ upd, const float* __restrict__ agg,
                      const float* __restrict__ g, const float* __restrict__ s, int M)
{
  size_t t = (size_t)blockIdx.x * blockDim.x + threadIdx.x;
  if (t >= (size_t)M * 128) return;
  float a = agg[t];
  upd[t] = a + g[t] * (s[t] - a);
}

// ---------------------------------------------------------------------------
// Host-side param table (JAX pytree order: dict keys sorted alphabetically)
// ---------------------------------------------------------------------------
struct Cur {
  void* const* din; int idx; const float* base; size_t off; bool sep;
  const float* next(size_t n) {
    if (sep) return (const float*)din[idx++];
    const float* p = base + off; off += n; return p;
  }
};

struct MlpP  { const float *b1,*b2,*ln_b,*ln_g,*w1,*w2; };
struct FourP { const float *b1,*b2,*freqs,*ln_b,*ln_g,*out_b,*out_ln_b,*out_ln_g,*out_w,*w1,*w2; };
struct AttnP {
  const float *bg,*bo,*bq,*bs,*bv,*bvr,*ff_b1,*ff_b2,*ff_ln_b,*ff_ln_g,*ff_w1,*ff_w2;
  const float *ln_dst_b,*ln_dst_g,*ln_r_b,*ln_r_g,*ln_src_b,*ln_src_g;
  const float *wg,*wk,*wkr,*wo,*wq,*ws,*wv,*wvr;
};

static MlpP readMlp(Cur& c, int fi, int fo) {
  MlpP m;
  m.b1 = c.next(128); m.b2 = c.next(fo); m.ln_b = c.next(128); m.ln_g = c.next(128);
  m.w1 = c.next((size_t)fi * 128); m.w2 = c.next((size_t)128 * fo);
  return m;
}
static FourP readFour(Cur& c, int C) {
  FourP f;
  f.b1 = c.next((size_t)C * 128); f.b2 = c.next((size_t)C * 128);
  f.freqs = c.next((size_t)C * 64);
  f.ln_b = c.next((size_t)C * 128); f.ln_g = c.next((size_t)C * 128);
  f.out_b = c.next(128); f.out_ln_b = c.next(128); f.out_ln_g = c.next(128);
  f.out_w = c.next(128 * 128);
  f.w1 = c.next((size_t)C * 129 * 128); f.w2 = c.next((size_t)C * 128 * 128);
  return f;
}
static AttnP readAttn(Cur& c) {
  AttnP a;
  a.bg = c.next(128); a.bo = c.next(128); a.bq = c.next(128); a.bs = c.next(128);
  a.bv = c.next(128); a.bvr = c.next(128);
  a.ff_b1 = c.next(512); a.ff_b2 = c.next(128);
  a.ff_ln_b = c.next(128); a.ff_ln_g = c.next(128);
  a.ff_w1 = c.next(128 * 512); a.ff_w2 = c.next(512 * 128);
  a.ln_dst_b = c.next(128); a.ln_dst_g = c.next(128);
  a.ln_r_b = c.next(128);   a.ln_r_g = c.next(128);
  a.ln_src_b = c.next(128); a.ln_src_g = c.next(128);
  a.wg = c.next(256 * 128); a.wk = c.next(128 * 128); a.wkr = c.next(128 * 128);
  a.wo = c.next(128 * 128); a.wq = c.next(128 * 128); a.ws = c.next(128 * 128);
  a.wv = c.next(128 * 128); a.wvr = c.next(128 * 128);
  return a;
}

// ---------------------------------------------------------------------------
extern "C" void kernel_launch(void* const* d_in, const int* in_sizes, int n_in,
                              void* d_out, int out_size, void* d_ws, size_t ws_size,
                              hipStream_t stream)
{
  const float* pos_a      = (const float*)d_in[0];
  const float* head_a     = (const float*)d_in[1];
  const float* agent_shp  = (const float*)d_in[2];
  const float* tok_veh    = (const float*)d_in[3];
  const float* tok_ped    = (const float*)d_in[4];
  const float* tok_cyc    = (const float*)d_in[5];
  const float* map_pos    = (const float*)d_in[6];
  const float* map_orient = (const float*)d_in[7];
  const float* map_feat   = (const float*)d_in[8];
  const int*   tok_idx    = (const int*)d_in[9];
  const int*   atype      = (const int*)d_in[10];
  const int*   e_t        = (const int*)d_in[11];
  const int*   e_a2a      = (const int*)d_in[12];
  const int*   e_pl2a     = (const int*)d_in[13];
  const int ET = in_sizes[11] / 2;
  const int EA = in_sizes[12] / 2;
  const int EP = in_sizes[13] / 2;
  const int EMAX = (EA > EP) ? EA : EP;

  // ---- params (sorted pytree order) ----
  Cur cur; cur.din = d_in; cur.idx = 14; cur.off = 0;
  cur.sep  = (n_in > 15);
  cur.base = (const float*)d_in[14];

  AttnP aL[6]; for (int i = 0; i < 6; ++i) aL[i] = readAttn(cur);
  MlpP fusionP = readMlp(cur, 256, 128);
  MlpP headP   = readMlp(cur, 128, NTOK);
  AttnP pL[6]; for (int i = 0; i < 6; ++i) pL[i] = readAttn(cur);
  FourP ra2aP  = readFour(cur, 3);
  FourP rpl2aP = readFour(cur, 3);
  FourP rtP    = readFour(cur, 4);
  MlpP shapeP  = readMlp(cur, 3, 128);
  AttnP tL[6]; for (int i = 0; i < 6; ++i) tL[i] = readAttn(cur);
  MlpP tokC = readMlp(cur, 8, 128);
  MlpP tokP = readMlp(cur, 8, 128);
  MlpP tokV = readMlp(cur, 8, 128);
  const float* type_table = cur.next(3 * 128);
  FourP xaP = readFour(cur, 2);

  // ---- workspace arena ----
  float* wsf = (float*)d_ws;
  size_t off = 0;
  auto alloc = [&](size_t n) { float* p = wsf + off; off += n; return p; };

  float* tables = alloc((size_t)3 * NTOK * 128);
  float* emb    = alloc((size_t)NSN * 128);
  float* xa     = alloc((size_t)NSN * 128);
  float* cat_a  = alloc((size_t)NAGT * 128);
  float* feat   = alloc((size_t)NSN * 128);
  float* fsbuf  = alloc((size_t)NSN * 128);
  float* featc  = alloc((size_t)NSN * 2);
  float* r_t    = alloc((size_t)ET * 128);
  float* r_a2a  = alloc((size_t)EA * 128);
  float* r_pl2a = alloc((size_t)EP * 128);
  float* relt4  = alloc((size_t)ET * 4);
  float* rela3  = alloc((size_t)EA * 3);
  float* relp3  = alloc((size_t)EP * 3);
  float* map_s  = alloc((size_t)NPLS * 128);
  float* lnS    = alloc((size_t)NPLS * 128);
  float* lnD    = alloc((size_t)NSN * 128);
  float* lnR    = alloc((size_t)EMAX * 128);
  float* qb     = alloc((size_t)NSN * 128);
  size_t scA = (size_t)EMAX * 132 + 64; if (scA < (size_t)NPLS * 128) scA = (size_t)NPLS * 128;
  size_t scB = (size_t)EMAX * 128; if (scB < (size_t)NPLS * 128) scB = (size_t)NPLS * 128;
  float* scratchA = alloc(scA);                 // fourier feat (stride 132) / k buffer
  float* scratchB = alloc(scB);                 // fourier hidden / v buffer / mlp hidden
  float* scratchC = alloc((size_t)EMAX * 128);  // fourier sum / rk buffer
  float* rv       = alloc((size_t)EMAX * 128);
  float* simb     = alloc((size_t)EMAX * 8);
  float* smax     = alloc((size_t)NSN * 8);
  float* den      = alloc((size_t)NSN * 8);
  float* agg      = alloc((size_t)NSN * 128);
  float* cat2     = alloc((size_t)NSN * 256);
  float* gbuf     = alloc((size_t)NSN * 128);
  float* sbuf     = alloc((size_t)NSN * 128);
  float* updb     = alloc((size_t)NSN * 128);
  float* ffh      = alloc((size_t)NSN * 512);
  __bf16* wbf     = (__bf16*)alloc(200000);     // preconverted weights (<=262144 bf16)
  (void)ws_size; (void)out_size;

  // ---- launch helpers ----
  auto gemm = [&](const float* A, const float* W, const float* bias, const float* add,
                  float* C, int M, int N, int K, int lda, int acc, int act, int safeK) {
    int KT = CDIV(K, 32);
    size_t nconv = (size_t)KT * 32 * N;
    k_w2bf<<<(unsigned)CDIV(nconv, (size_t)256), 256, 0, stream>>>(W, wbf, K, N, KT);
    dim3 g(CDIV(M, 64), N / 128);
    k_gemm<<<g, 256, 0, stream>>>(A, wbf, bias, add, C, M, N, K, KT, lda, acc, act, safeK);
  };
  auto ln = [&](const float* x, const float* g_, const float* b_, float* y, int M, int relu) {
    k_ln128<<<CDIV(M, 8), 256, 0, stream>>>(x, g_, b_, y, M, relu);
  };
  auto fill = [&](float* p, float v, size_t n) {
    k_fill<<<(unsigned)CDIV(n, (size_t)256), 256, 0, stream>>>(p, v, n);
  };
  auto mlp = [&](const float* x, const MlpP& P, int M, int fi, int fo, float* hbuf, float* out) {
    gemm(x, P.w1, P.b1, nullptr, hbuf, M, 128, fi, fi, 0, 0, 0);
    ln(hbuf, P.ln_g, P.ln_b, hbuf, M, 1);
    gemm(hbuf, P.w2, P.b2, nullptr, out, M, fo, 128, 128, 0, 0, 0);
  };
  auto fourier = [&](const float* x, int N, int C, const FourP& P, const float* catA, float* dst) {
    for (int c = 0; c < C; ++c) {
      k_ffeat<<<(unsigned)CDIV((size_t)N * 132, (size_t)256), 256, 0, stream>>>(
          x, C, c, P.freqs + (size_t)c * 64, scratchA, N);
      gemm(scratchA, P.w1 + (size_t)c * 129 * 128, P.b1 + (size_t)c * 128, nullptr,
           scratchB, N, 128, 129, 132, 0, 0, 1);
      ln(scratchB, P.ln_g + (size_t)c * 128, P.ln_b + (size_t)c * 128, scratchB, N, 1);
      gemm(scratchB, P.w2 + (size_t)c * 128 * 128, P.b2 + (size_t)c * 128, nullptr,
           scratchC, N, 128, 128, 128, (c > 0) ? 1 : 0, 0, 0);
    }
    if (catA)
      k_addagent<<<(unsigned)CDIV((size_t)N * 128, (size_t)256), 256, 0, stream>>>(scratchC, catA, N);
    ln(scratchC, P.out_ln_g, P.out_ln_b, scratchC, N, 1);
    gemm(scratchC, P.out_w, P.out_b, nullptr, dst, N, 128, 128, 128, 0, 0, 0);
  };
  auto attn = [&](const float* xsrc, float* xdst, const float* r,
                  const int* esrc, const int* edst, int nsrc, int ndst, int E,
                  const AttnP& P, bool bip) {
    float* kb = scratchA;
    float* vb = scratchB;
    float* rk = scratchC;
    ln(xsrc, P.ln_src_g, P.ln_src_b, lnS, nsrc, 0);
    const float* lnDp = lnS;
    if (bip) { ln(xdst, P.ln_dst_g, P.ln_dst_b, lnD, ndst, 0); lnDp = lnD; }
    ln(r, P.ln_r_g, P.ln_r_b, lnR, E, 0);
    gemm(lnDp, P.wq, P.bq, nullptr, qb, ndst, 128, 128, 128, 0, 0, 0);
    gemm(lnS, P.wk, nullptr, nullptr, kb, nsrc, 128, 128, 128, 0, 0, 0);
    gemm(lnS, P.wv, P.bv, nullptr, vb, nsrc, 128, 128, 128, 0, 0, 0);
    gemm(lnR, P.wkr, nullptr, nullptr, rk, E, 128, 128, 128, 0, 0, 0);
    gemm(lnR, P.wvr, P.bvr, nullptr, rv, E, 128, 128, 128, 0, 0, 0);
    k_sim<<<(unsigned)CDIV((size_t)E * 8, (size_t)256), 256, 0, stream>>>(qb, kb, rk, esrc, edst, simb, E);
    fill(smax, -1e30f, (size_t)ndst * 8);
    fill(den, 0.f, (size_t)ndst * 8);
    fill(agg, 0.f, (size_t)ndst * 128);
    k_segmax<<<(unsigned)CDIV((size_t)E * 8, (size_t)256), 256, 0, stream>>>(simb, edst, smax, E);
    k_expden<<<(unsigned)CDIV((size_t)E * 8, (size_t)256), 256, 0, stream>>>(simb, edst, smax, den, E);
    k_agg<<<(unsigned)CDIV((size_t)E * 128, (size_t)256), 256, 0, stream>>>(simb, den, vb, rv, esrc, edst, agg, E);
    k_concat2<<<(unsigned)CDIV((size_t)ndst * 256, (size_t)256), 256, 0, stream>>>(cat2, agg, lnDp, ndst);
    gemm(cat2, P.wg, P.bg, nullptr, gbuf, ndst, 128, 256, 256, 0, 2, 0);
    gemm(lnDp, P.ws, P.bs, nullptr, sbuf, ndst, 128, 128, 128, 0, 0, 0);
    k_upd<<<(unsigned)CDIV((size_t)ndst * 128, (size_t)256), 256, 0, stream>>>(updb, agg, gbuf, sbuf, ndst);
    gemm(updb, P.wo, P.bo, xdst, xdst, ndst, 128, 128, 128, 0, 0, 0);
    ln(xdst, P.ff_ln_g, P.ff_ln_b, sbuf, ndst, 0);
    gemm(sbuf, P.ff_w1, P.ff_b1, nullptr, ffh, ndst, 512, 128, 128, 0, 1, 0);
    gemm(ffh, P.ff_w2, P.ff_b2, xdst, xdst, ndst, 128, 512, 512, 0, 0, 0);
  };

  // ---- 1. token tables + embedding gather ----
  mlp(tok_veh, tokV, NTOK, 8, 128, scratchB, tables + (size_t)0 * NTOK * 128);
  mlp(tok_ped, tokP, NTOK, 8, 128, scratchB, tables + (size_t)1 * NTOK * 128);
  mlp(tok_cyc, tokC, NTOK, 8, 128, scratchB, tables + (size_t)2 * NTOK * 128);
  k_gather_emb<<<(unsigned)CDIV((size_t)NSN * 128, (size_t)256), 256, 0, stream>>>(tables, tok_idx, atype, emb);

  // ---- 2. categorical sum (type table + shape MLP) ----
  mlp(agent_shp, shapeP, NAGT, 3, 128, scratchB, cat_a);
  k_typeadd<<<(unsigned)CDIV((size_t)NAGT * 128, (size_t)256), 256, 0, stream>>>(cat_a, type_table, atype);

  // ---- 3. continuous motion features -> fourier x_a ----
  k_featcont<<<CDIV(NSN, 256), 256, 0, stream>>>(pos_a, head_a, featc);
  fourier(featc, NSN, 2, xaP, cat_a, xa);

  // ---- 4. fusion MLP ----
  k_concat2<<<(unsigned)CDIV((size_t)NSN * 256, (size_t)256), 256, 0, stream>>>(cat2, emb, xa, NSN);
  mlp(cat2, fusionP, NSN, 256, 128, scratchB, feat);

  // ---- 5. relation embeddings ----
  k_rel_t<<<CDIV(ET, 256), 256, 0, stream>>>(pos_a, head_a, e_t, e_t + ET, relt4, ET);
  k_rel_a2a<<<CDIV(EA, 256), 256, 0, stream>>>(pos_a, head_a, e_a2a, e_a2a + EA, rela3, EA);
  k_rel_pl2a<<<CDIV(EP, 256), 256, 0, stream>>>(map_pos, map_orient, pos_a, head_a,
                                                e_pl2a, e_pl2a + EP, relp3, EP);
  fourier(relt4, ET, 4, rtP, nullptr, r_t);
  fourier(rela3, EA, 3, ra2aP, nullptr, r_a2a);
  fourier(relp3, EP, 3, rpl2aP, nullptr, r_pl2a);

  // ---- 6. tiled map features (constant across layers) ----
  k_tile_map<<<(unsigned)CDIV((size_t)NPLS * 128, (size_t)256), 256, 0, stream>>>(map_s, map_feat);

  // ---- 7. attention stack ----
  for (int i = 0; i < 6; ++i) {
    attn(feat, feat, r_t, e_t, e_t + ET, NSN, NSN, ET, tL[i], false);
    k_transpose<<<(unsigned)CDIV((size_t)NSN * 128, (size_t)256), 256, 0, stream>>>(fsbuf, feat, 0);
    attn(map_s, fsbuf, r_pl2a, e_pl2a, e_pl2a + EP, NPLS, NSN, EP, pL[i], true);
    attn(fsbuf, fsbuf, r_a2a, e_a2a, e_a2a + EA, NSN, NSN, EA, aL[i], false);
    k_transpose<<<(unsigned)CDIV((size_t)NSN * 128, (size_t)256), 256, 0, stream>>>(feat, fsbuf, 1);
  }

  // ---- 8. head ----
  mlp(feat, headP, NSN, 128, NTOK, sbuf, (float*)d_out);
}
